// EfficientDet_28681791603113
// MI455X (gfx1250) — matrix operand order repaired and checked by reading, not compiled, not run on hardware
//
#include <hip/hip_runtime.h>
#include <hip/hip_fp16.h>
#include <math.h>

typedef __attribute__((ext_vector_type(16))) _Float16 v16h;
typedef __attribute__((ext_vector_type(8)))  _Float16 v8h;
typedef __attribute__((ext_vector_type(8)))  float    v8f;

#define MODE_PLAIN    0
#define MODE_RELU     1
#define MODE_HEAD     2
#define MODE_HEAD_SIG 3

// ---------------------------------------------------------------------------
// Implicit-GEMM conv (KS=1 pointwise or KS=3 SAME) via v_wmma_f32_16x16x32_f16.
// Block = 256 threads = 8 waves; block tile = 32 cout x 128 pixels.
// Each wave owns one 16-pixel N-tile and BOTH 16-row M-tiles (two f32 acc
// chains sharing one B fragment). A tile [32co x 32k] and B tile [128n x 32k]
// (k-contiguous, row stride 40 halves = 80B so every fragment piece is a
// 16B-aligned ds_load_b128, bank-conflict-free). Epilogue fuses bias(+ReLU)
// or the head transpose (+sigmoid) into [B, Atot, oc].
// ---------------------------------------------------------------------------
template <int KS, int MODE>
__global__ void __launch_bounds__(256)
conv_wmma_kernel(const float* __restrict__ X, const float* __restrict__ Wt,
                 const float* __restrict__ Bias, float* __restrict__ Y,
                 int Cin, int Cout, int H, int Wd,
                 int oc, int anchorOff, int Atot)
{
  constexpr int KK = KS * KS;
  __shared__ __align__(16) _Float16 sA[32 * 40];    // [co][k], stride 40
  __shared__ __align__(16) _Float16 sBt[128 * 40];  // [n][k],  stride 40

  const int tid  = threadIdx.x;
  const int lane = tid & 31;
  const int wave = tid >> 5;          // N sub-tile 0..7
  const int N    = H * Wd;
  const int cotile = blockIdx.x * 32;
  const int ntile  = blockIdx.y * 128;
  const int b      = blockIdx.z;
  const size_t xbase = (size_t)b * Cin * N;

  // ---- cooperative-load invariants (computed once) ----
  const int aci  = tid & 31;   // A: k column
  const int arow = tid >> 5;   // A: co row base (step 8)
  const int bcol = tid & 127;  // B: n column
  const int brow = tid >> 7;   // B: k row base (step 2)
  const int nB   = ntile + bcol;
  const bool nok = nB < N;
  const int py   = nok ? (nB / Wd) : 0;
  const int px   = nB - py * Wd;

  // ---- fragment-build invariants ----
  const int hsel = (lane < 16) ? 0 : 1;
  const int mr   = lane & 15;
  const int kbA  = hsel * 8;                 // A halves: {0..7,16..23} / {8..15,24..31}
  const int kbB  = hsel * 8;                 // B halves: {0..7,16..23} / {8..15,24..31}
  const _Float16* pA0 = &sA[mr * 40 + kbA];
  const _Float16* pA1 = &sA[(mr + 16) * 40 + kbA];
  const _Float16* pB  = &sBt[(wave * 16 + mr) * 40 + kbB];

  v8f acc0 = {}, acc1 = {};

  for (int tap = 0; tap < KK; ++tap) {
    const int dy = tap / KS - KS / 2;
    const int dx = tap % KS - KS / 2;
    const int yy = py + dy, xx = px + dx;
    const bool valid = nok && yy >= 0 && yy < H && xx >= 0 && xx < Wd;
    const size_t src = xbase + (size_t)yy * Wd + xx;

    for (int k0 = 0; k0 < Cin; k0 += 32) {
      // ---- stage A tile (weights): rows co = cotile + arow + 8i ----
      {
        const int ci = k0 + aci;
        const bool ciok = ci < Cin;
        #pragma unroll
        for (int i = 0; i < 4; ++i) {
          int r = arow + i * 8;
          int co = cotile + r;
          float v = (ciok && co < Cout) ? Wt[(size_t)(co * Cin + ci) * KK + tap] : 0.f;
          sA[r * 40 + aci] = (_Float16)v;
        }
      }
      // ---- stage B tile (shifted pixels): rows k = brow + 2i, col n fixed ----
      #pragma unroll
      for (int i = 0; i < 16; ++i) {
        int r = brow + i * 2;
        int ci = k0 + r;
        float v = (valid && ci < Cin) ? X[src + (size_t)ci * N] : 0.f;
        sBt[bcol * 40 + r] = (_Float16)v;
      }
      __syncthreads();

      // ---- fragments: all 16B-aligned ds_load_b128 ----
      v8h a0lo = *(const v8h*)(pA0);
      v8h a0hi = *(const v8h*)(pA0 + 16);
      v8h a1lo = *(const v8h*)(pA1);
      v8h a1hi = *(const v8h*)(pA1 + 16);
      v8h blo  = *(const v8h*)(pB);
      v8h bhi  = *(const v8h*)(pB + 16);
      v16h af0 = __builtin_shufflevector(a0lo, a0hi, 0,1,2,3,4,5,6,7,8,9,10,11,12,13,14,15);
      v16h af1 = __builtin_shufflevector(a1lo, a1hi, 0,1,2,3,4,5,6,7,8,9,10,11,12,13,14,15);
      v16h bf  = __builtin_shufflevector(blo,  bhi,  0,1,2,3,4,5,6,7,8,9,10,11,12,13,14,15);

      acc0 = __builtin_amdgcn_wmma_f32_16x16x32_f16(false, af0, false, bf,
                                                    (short)0, acc0, false, false);
      acc1 = __builtin_amdgcn_wmma_f32_16x16x32_f16(false, af1, false, bf,
                                                    (short)0, acc1, false, false);
      __syncthreads();
    }
  }

  // ---- epilogue: two M groups per wave ----
  const int nc = ntile + wave * 16 + mr;
  if (nc < N) {
    #pragma unroll
    for (int g = 0; g < 2; ++g) {
      const v8f& acc = g ? acc1 : acc0;
      const int mbase = cotile + g * 16 + ((lane < 16) ? 0 : 8);
      #pragma unroll
      for (int j = 0; j < 8; ++j) {
        int m = mbase + j;
        if (m < Cout) {
          float v = acc[j] + Bias[m];
          if (MODE == MODE_RELU) v = fmaxf(v, 0.f);
          if (MODE == MODE_PLAIN || MODE == MODE_RELU) {
            Y[((size_t)b * Cout + m) * N + nc] = v;
          } else {
            if (MODE == MODE_HEAD_SIG) v = 1.f / (1.f + __expf(-v));
            int a = m / oc, c = m - a * oc;   // channel = a*oc + c
            Y[((size_t)b * Atot + anchorOff + nc * 9 + a) * oc + c] = v;
          }
        }
      }
    }
  }
}

// ---------------------------------------------------------------------------
// Depthwise 3x3 SAME (memory-bound, VALU).
// ---------------------------------------------------------------------------
__global__ void dwconv3x3_kernel(const float* __restrict__ X,
                                 const float* __restrict__ Wt,
                                 float* __restrict__ Y,
                                 int C, int H, int W, int total)
{
  int idx = blockIdx.x * blockDim.x + threadIdx.x;
  if (idx >= total) return;
  int x = idx % W; int t = idx / W;
  int y = t % H;   t /= H;
  int c = t % C;
  const float* w  = Wt + c * 9;
  const float* xp = X + (size_t)(idx - (y * W + x));  // (b,c) plane base
  float s = 0.f;
  #pragma unroll
  for (int ky = 0; ky < 3; ++ky) {
    int yy = y + ky - 1;
    if (yy < 0 || yy >= H) continue;
    #pragma unroll
    for (int kx = 0; kx < 3; ++kx) {
      int xx = x + kx - 1;
      if (xx < 0 || xx >= W) continue;
      s += w[ky * 3 + kx] * xp[yy * W + xx];
    }
  }
  Y[idx] = s;
}

// ---------------------------------------------------------------------------
// Fast-normalized fusion variants (up2 / down2 folded in).
// ---------------------------------------------------------------------------
__global__ void fuse2_up_kernel(const float* __restrict__ fw,
                                const float* __restrict__ A,
                                const float* __restrict__ Bs,  // half-res
                                float* __restrict__ O, int H, int W, int total)
{
  int idx = blockIdx.x * blockDim.x + threadIdx.x;
  if (idx >= total) return;
  float w0 = fmaxf(fw[0], 0.f), w1 = fmaxf(fw[1], 0.f);
  float inv = 1.f / (w0 + w1 + 1e-4f);
  int x = idx % W; int t = idx / W; int y = t % H; int pc = t / H;
  float bv = Bs[((size_t)pc * (H >> 1) + (y >> 1)) * (W >> 1) + (x >> 1)];
  O[idx] = (w0 * A[idx] + w1 * bv) * inv;
}

__global__ void fuse2_down_kernel(const float* __restrict__ fw,
                                  const float* __restrict__ A,
                                  const float* __restrict__ Big, // 2H x 2W
                                  float* __restrict__ O, int H, int W, int total)
{
  int idx = blockIdx.x * blockDim.x + threadIdx.x;
  if (idx >= total) return;
  float w0 = fmaxf(fw[0], 0.f), w1 = fmaxf(fw[1], 0.f);
  float inv = 1.f / (w0 + w1 + 1e-4f);
  int x = idx % W; int t = idx / W; int y = t % H; int pc = t / H;
  const float* bp = Big + ((size_t)pc * (2 * H) + 2 * y) * (2 * W) + 2 * x;
  float m = fmaxf(fmaxf(bp[0], bp[1]), fmaxf(bp[2 * W], bp[2 * W + 1]));
  O[idx] = (w0 * A[idx] + w1 * m) * inv;
}

__global__ void fuse3_down_kernel(const float* __restrict__ fw,
                                  const float* __restrict__ A,
                                  const float* __restrict__ Bm,  // same res
                                  const float* __restrict__ Big, // 2H x 2W
                                  float* __restrict__ O, int H, int W, int total)
{
  int idx = blockIdx.x * blockDim.x + threadIdx.x;
  if (idx >= total) return;
  float w0 = fmaxf(fw[0], 0.f), w1 = fmaxf(fw[1], 0.f), w2 = fmaxf(fw[2], 0.f);
  float inv = 1.f / (w0 + w1 + w2 + 1e-4f);
  int x = idx % W; int t = idx / W; int y = t % H; int pc = t / H;
  const float* bp = Big + ((size_t)pc * (2 * H) + 2 * y) * (2 * W) + 2 * x;
  float m = fmaxf(fmaxf(bp[0], bp[1]), fmaxf(bp[2 * W], bp[2 * W + 1]));
  O[idx] = (w0 * A[idx] + w1 * Bm[idx] + w2 * m) * inv;
}

// ---------------------------------------------------------------------------
// RetinaNet-style anchors for one pyramid level.
// ---------------------------------------------------------------------------
__global__ void anchors_kernel(float* __restrict__ out, int fs, int stride,
                               int size, int off, int total)
{
  int idx = blockIdx.x * blockDim.x + threadIdx.x;
  if (idx >= total) return;
  int k = idx % 9; int t = idx / 9;
  int ix = t % fs; int iy = t / fs;
  int ri = k / 3, si = k % 3;
  float scale = exp2f((float)si * (1.f / 3.f));
  float ratio = (ri == 0) ? 0.5f : ((ri == 1) ? 1.f : 2.f);
  float sr = sqrtf(ratio);
  float w = (float)size * scale / sr;
  float h = (float)size * scale * sr;
  float cx = (ix + 0.5f) * (float)stride;
  float cy = (iy + 0.5f) * (float)stride;
  float* o = out + (size_t)(off + idx) * 4;
  o[0] = cx - 0.5f * w; o[1] = cy - 0.5f * h;
  o[2] = cx + 0.5f * w; o[3] = cy + 0.5f * h;
}

// ---------------------------------------------------------------------------
// Host orchestration
// ---------------------------------------------------------------------------
extern "C" void kernel_launch(void* const* d_in, const int* in_sizes, int n_in,
                              void* d_out, int out_size, void* d_ws, size_t ws_size,
                              hipStream_t stream)
{
  (void)in_sizes; (void)n_in; (void)out_size; (void)ws_size;
  const int B = 4, C = 90, ST = 3, D = 4, NCLS = 90;
  const int Hs[5]    = {64, 32, 16, 8, 4};
  const int chans[5] = {40, 80, 112, 192, 320};

  const float *c_in[5], *latw[5], *latb[5];
  for (int i = 0; i < 5; ++i) {
    c_in[i] = (const float*)d_in[3 * i];
    latw[i] = (const float*)d_in[3 * i + 1];
    latb[i] = (const float*)d_in[3 * i + 2];
  }
  const float* bif_dw = (const float*)d_in[15];
  const float* bif_pw = (const float*)d_in[16];
  const float* bif_pb = (const float*)d_in[17];
  const float* fw2    = (const float*)d_in[18];
  const float* fw3    = (const float*)d_in[19];
  const float* reg_tw = (const float*)d_in[20];
  const float* reg_tb = (const float*)d_in[21];
  const float* reg_ow = (const float*)d_in[22];
  const float* reg_ob = (const float*)d_in[23];
  const float* cls_tw = (const float*)d_in[24];
  const float* cls_tb = (const float*)d_in[25];
  const float* cls_ow = (const float*)d_in[26];
  const float* cls_ob = (const float*)d_in[27];

  int HW[5], aoff[5], Atot = 0;
  for (int i = 0; i < 5; ++i) { HW[i] = Hs[i] * Hs[i]; aoff[i] = Atot; Atot += HW[i] * 9; }

  float* cls_out = (float*)d_out;                          // [B, Atot, 90]
  float* reg_out = cls_out + (size_t)B * Atot * NCLS;      // [B, Atot, 4]
  float* anc_out = reg_out + (size_t)B * Atot * 4;         // [Atot, 4]

  // workspace carve-out
  float* ws = (float*)d_ws;
  float *P[5], *Q[5], *Tt[5];
  size_t off = 0;
  for (int i = 0; i < 5; ++i) { P[i] = ws + off; off += (size_t)B * C * HW[i]; }
  for (int i = 0; i < 5; ++i) { Q[i] = ws + off; off += (size_t)B * C * HW[i]; }
  for (int i = 1; i <= 3; ++i) { Tt[i] = ws + off; off += (size_t)B * C * HW[i]; }
  float* F    = ws + off; off += (size_t)B * C * HW[0];
  float* Dw   = ws + off; off += (size_t)B * C * HW[0];
  float* Tow0 = ws + off; off += (size_t)B * C * HW[0];
  float* Tow1 = ws + off; off += (size_t)B * C * HW[0];

  dim3 blk(256);
  auto cgrid = [&](int Cout, int N) { return dim3((Cout + 31) / 32, (N + 127) / 128, B); };

  auto c1_plain = [&](const float* X, const float* Wt, const float* Bi, float* Y,
                      int Cin, int Cout, int H, int W) {
    conv_wmma_kernel<1, MODE_PLAIN><<<cgrid(Cout, H * W), blk, 0, stream>>>(
        X, Wt, Bi, Y, Cin, Cout, H, W, 1, 0, 0);
  };
  auto c1_relu = [&](const float* X, const float* Wt, const float* Bi, float* Y,
                     int Cin, int Cout, int H, int W) {
    conv_wmma_kernel<1, MODE_RELU><<<cgrid(Cout, H * W), blk, 0, stream>>>(
        X, Wt, Bi, Y, Cin, Cout, H, W, 1, 0, 0);
  };
  auto c3_relu = [&](const float* X, const float* Wt, const float* Bi, float* Y,
                     int Cin, int Cout, int H, int W) {
    conv_wmma_kernel<3, MODE_RELU><<<cgrid(Cout, H * W), blk, 0, stream>>>(
        X, Wt, Bi, Y, Cin, Cout, H, W, 1, 0, 0);
  };
  auto c3_head = [&](const float* X, const float* Wt, const float* Bi, float* Y,
                     int Cin, int Cout, int H, int W, int oc, int aofs) {
    conv_wmma_kernel<3, MODE_HEAD><<<cgrid(Cout, H * W), blk, 0, stream>>>(
        X, Wt, Bi, Y, Cin, Cout, H, W, oc, aofs, Atot);
  };
  auto c3_headsig = [&](const float* X, const float* Wt, const float* Bi, float* Y,
                        int Cin, int Cout, int H, int W, int oc, int aofs) {
    conv_wmma_kernel<3, MODE_HEAD_SIG><<<cgrid(Cout, H * W), blk, 0, stream>>>(
        X, Wt, Bi, Y, Cin, Cout, H, W, oc, aofs, Atot);
  };

  // ---- lateral 1x1 convs ----
  for (int i = 0; i < 5; ++i)
    c1_plain(c_in[i], latw[i], latb[i], P[i], chans[i], C, Hs[i], Hs[i]);

  // ---- BiFPN layers ----
  for (int s = 0; s < ST; ++s) {
    const float* dwp = bif_dw + (size_t)(s * 8) * C * 9;
    const float* pwp = bif_pw + (size_t)(s * 8) * C * C;
    const float* pbp = bif_pb + (size_t)(s * 8) * C;

    auto sep = [&](int i, int lvl, float* dst) {
      int total = B * C * HW[lvl];
      dwconv3x3_kernel<<<(total + 255) / 256, blk, 0, stream>>>(
          F, dwp + (size_t)i * C * 9, Dw, C, Hs[lvl], Hs[lvl], total);
      c1_relu(Dw, pwp + (size_t)i * C * C, pbp + (size_t)i * C, dst,
              C, C, Hs[lvl], Hs[lvl]);
    };
    auto f2up = [&](int row, int lvl, const float* a, const float* bsm) {
      int total = B * C * HW[lvl];
      fuse2_up_kernel<<<(total + 255) / 256, blk, 0, stream>>>(
          fw2 + (size_t)(s * 5 + row) * 2, a, bsm, F, Hs[lvl], Hs[lvl], total);
    };
    auto f2dn = [&](int row, int lvl, const float* a, const float* big) {
      int total = B * C * HW[lvl];
      fuse2_down_kernel<<<(total + 255) / 256, blk, 0, stream>>>(
          fw2 + (size_t)(s * 5 + row) * 2, a, big, F, Hs[lvl], Hs[lvl], total);
    };
    auto f3dn = [&](int row, int lvl, const float* a, const float* bm, const float* big) {
      int total = B * C * HW[lvl];
      fuse3_down_kernel<<<(total + 255) / 256, blk, 0, stream>>>(
          fw3 + (size_t)(s * 3 + row) * 3, a, bm, big, F, Hs[lvl], Hs[lvl], total);
    };

    f2up(0, 3, P[3], P[4]);        sep(0, 3, Tt[3]);   // p6t
    f2up(1, 2, P[2], Tt[3]);       sep(1, 2, Tt[2]);   // p5t
    f2up(2, 1, P[1], Tt[2]);       sep(2, 1, Tt[1]);   // p4t
    f2up(3, 0, P[0], Tt[1]);       sep(3, 0, Q[0]);    // p3o
    f3dn(0, 1, P[1], Tt[1], Q[0]); sep(4, 1, Q[1]);    // p4o
    f3dn(1, 2, P[2], Tt[2], Q[1]); sep(5, 2, Q[2]);    // p5o
    f3dn(2, 3, P[3], Tt[3], Q[2]); sep(6, 3, Q[3]);    // p6o
    f2dn(4, 4, P[4], Q[3]);        sep(7, 4, Q[4]);    // p7o
    for (int i = 0; i < 5; ++i) { float* t = P[i]; P[i] = Q[i]; Q[i] = t; }
  }

  // ---- heads (tower convs + fused transpose/sigmoid output) ----
  float* pp[2] = {Tow0, Tow1};
  for (int lvl = 0; lvl < 5; ++lvl) {
    const float* x = P[lvl];
    for (int i = 0; i < D; ++i) {
      c3_relu(x, reg_tw + (size_t)i * C * C * 9, reg_tb + (size_t)i * C,
              pp[i & 1], C, C, Hs[lvl], Hs[lvl]);
      x = pp[i & 1];
    }
    c3_head(x, reg_ow, reg_ob, reg_out, C, 36, Hs[lvl], Hs[lvl], 4, aoff[lvl]);

    x = P[lvl];
    for (int i = 0; i < D; ++i) {
      c3_relu(x, cls_tw + (size_t)i * C * C * 9, cls_tb + (size_t)i * C,
              pp[i & 1], C, C, Hs[lvl], Hs[lvl]);
      x = pp[i & 1];
    }
    c3_headsig(x, cls_ow, cls_ob, cls_out, C, 9 * NCLS, Hs[lvl], Hs[lvl],
               NCLS, aoff[lvl]);
  }

  // ---- anchors ----
  for (int lvl = 0; lvl < 5; ++lvl) {
    int strd = 1 << (lvl + 3);      // 2^(lvl+3)
    int size = 1 << (lvl + 5);      // 2^(lvl+5)
    int total = HW[lvl] * 9;
    anchors_kernel<<<(total + 255) / 256, blk, 0, stream>>>(
        anc_out, Hs[lvl], strd, size, aoff[lvl], total);
  }
}
